// srnn_70308614636173
// MI455X (gfx1250) — matrix operand order repaired and checked by reading, not compiled, not run on hardware
//
#include <hip/hip_runtime.h>

// MI455X (gfx1250) fused SRNN scan.
//   h_{t+1} = tanh(A h_t + Bm u_t + C),  out = D.h + E
// 4 workgroups (one per 16-batch tile), 256 threads = 8 wave32s each.
// Transposed WMMA product: D-tile[m=h_out][n=batch]; A/Bm/C/D fragments are
// hoisted into VGPRs (loop-invariant); h state double-buffered in LDS as bf16;
// one barrier per timestep; output dot fused via shfl + ds_add_f32.
// v2: all 8 h-fragments get distinct registers and are loaded up-front each
// step so the 16 ds_load_b128 overlap the u-projection WMMAs instead of
// stalling on s_wait_dscnt 0x0 every two WMMAs.

#define T_LEN 4096
#define H_DIM 256
#define I_DIM 64
#define B_DIM 64
#define HPAD  264   // padded LDS row stride (bf16 elems) -> conflict-free banks

typedef __attribute__((ext_vector_type(16))) __bf16 v16bf;
typedef __attribute__((ext_vector_type(8)))  __bf16 v8bf;
typedef __attribute__((ext_vector_type(8)))  float  v8f;

static __device__ __forceinline__ v8f wmma_bf16(v16bf a, v16bf b, v8f c) {
  // D = A(16x32 bf16) x B(32x16 bf16) + C(16x16 f32)
  return __builtin_amdgcn_wmma_f32_16x16x32_bf16(false, a, false, b, (short)0, c,
                                                 false, false);
}

static __device__ __forceinline__ float fast_tanh(float x) {
#if __has_builtin(__builtin_amdgcn_tanhf)
  return __builtin_amdgcn_tanhf(x);          // v_tanh_f32 (gfx1250 TRANS op)
#else
  float r;
  asm("v_tanh_f32 %0, %1" : "=v"(r) : "v"(x));
  return r;
#endif
}

// load 8 contiguous f32 (32B aligned) -> 8 bf16
static __device__ __forceinline__ v8bf ldcvt8(const float* __restrict__ p) {
  const float4* q = (const float4*)p;
  float4 a = q[0];
  float4 b = q[1];
  v8bf r;
  r[0] = (__bf16)a.x; r[1] = (__bf16)a.y; r[2] = (__bf16)a.z; r[3] = (__bf16)a.w;
  r[4] = (__bf16)b.x; r[5] = (__bf16)b.y; r[6] = (__bf16)b.z; r[7] = (__bf16)b.w;
  return r;
}

static __device__ __forceinline__ v16bf cat16(v8bf lo, v8bf hi) {
  return __builtin_shufflevector(lo, hi, 0,1,2,3,4,5,6,7,8,9,10,11,12,13,14,15);
}

static __device__ __forceinline__ v8f ld8f(const float* __restrict__ p) {
  const float4* q = (const float4*)p;
  float4 a = q[0];
  float4 b = q[1];
  v8f r;
  r[0] = a.x; r[1] = a.y; r[2] = a.z; r[3] = a.w;
  r[4] = b.x; r[5] = b.y; r[6] = b.z; r[7] = b.w;
  return r;
}

__global__ __launch_bounds__(256, 2) void srnn_fused_kernel(
    const float* __restrict__ u,   // [B, T, I]
    const float* __restrict__ A,   // [H, H]  (row = h_out, col = h_in)
    const float* __restrict__ Bm,  // [H, I]
    const float* __restrict__ C,   // [H]
    const float* __restrict__ D,   // [O=1, H]
    const float* __restrict__ E,   // [O=1]
    float* __restrict__ out)       // [B, T]
{
  __shared__ __align__(16) __bf16 hbuf[2][16][HPAD];  // double-buffered state
  __shared__ float sOut[2][16];                       // per-batch output accum

  const int tid  = threadIdx.x;
  const int lane = tid & 31;
  const int wv   = tid >> 5;     // wave 0..7
  const int lm   = lane & 15;    // N index (batch) / M row within tile
  const int hs   = lane >> 4;    // lane half select (WMMA layout)
  const int bt   = blockIdx.x;   // batch tile 0..3

  // zero h0 and output accumulators
  for (int i = tid; i < 2 * 16 * HPAD; i += 256) ((__bf16*)hbuf)[i] = (__bf16)0.0f;
  if (tid < 32) ((float*)sOut)[tid] = 0.0f;

  // each wave owns 2 M-tiles (h_out ranges of 16)
  const int mt0 = wv * 2, mt1 = wv * 2 + 1;
  const int row0 = mt0 * 16 + lm, row1 = mt1 * 16 + lm;

  // ---- hoist loop-invariant fragments into VGPRs ----
  // A-operand k order used here (contiguous, B matches): lane holds M=lm, K chunk
  // {base..base+15}, base = hs*16, per 32-wide K tile.
  v16bf afr0[8], afr1[8];
#pragma unroll
  for (int k = 0; k < 8; ++k) {
    const int col = k * 32 + hs * 16;
    afr0[k] = cat16(ldcvt8(A + row0 * H_DIM + col), ldcvt8(A + row0 * H_DIM + col + 8));
    afr1[k] = cat16(ldcvt8(A + row1 * H_DIM + col), ldcvt8(A + row1 * H_DIM + col + 8));
  }
  v16bf bmf0[2], bmf1[2];
#pragma unroll
  for (int k = 0; k < 2; ++k) {
    const int col = k * 32 + hs * 16;
    bmf0[k] = cat16(ldcvt8(Bm + row0 * I_DIM + col), ldcvt8(Bm + row0 * I_DIM + col + 8));
    bmf1[k] = cat16(ldcvt8(Bm + row1 * I_DIM + col), ldcvt8(Bm + row1 * I_DIM + col + 8));
  }
  // C/D fragments: f32 D-layout element r -> M = mt*16 + hs*8 + r
  const v8f cf0 = ld8f(C + mt0 * 16 + hs * 8);
  const v8f cf1 = ld8f(C + mt1 * 16 + hs * 8);
  const v8f df0 = ld8f(D + mt0 * 16 + hs * 8);
  const v8f df1 = ld8f(D + mt1 * 16 + hs * 8);
  const float E0 = E[0];

  // per-lane u stream pointer: batch = bt*16 + lm, B-operand K offset = hs*16
  const float* up = u + (size_t)(bt * 16 + lm) * (T_LEN * I_DIM) + hs * 16;

  __syncthreads();

  int cur = 0;
  for (int t = 0; t < T_LEN; ++t) {
    // ---- issue ALL state loads first: 16 ds_load_b128 into 8 distinct
    // fragments, so their latency overlaps the u-projection below ----
    const __bf16* hrow = &hbuf[cur][lm][hs * 16];
    v16bf hf[8];
#pragma unroll
    for (int k = 0; k < 8; ++k) {
      hf[k] = cat16(*(const v8bf*)(hrow + k * 32),
                    *(const v8bf*)(hrow + k * 32 + 8));
    }

    __builtin_prefetch(up + 8 * I_DIM, 0, 1);   // stream u ~8 steps ahead

    // u B-fragments (K = I = 64 -> two 32-wide K tiles), f32 -> bf16
    v16bf uf0 = cat16(ldcvt8(up),      ldcvt8(up + 8));
    v16bf uf1 = cat16(ldcvt8(up + 32), ldcvt8(up + 40));
    up += I_DIM;

    // acc = C, then += Bm @ u_t  (input projection fused)
    v8f acc0 = cf0, acc1 = cf1;
    acc0 = wmma_bf16(bmf0[0], uf0, acc0);
    acc1 = wmma_bf16(bmf1[0], uf0, acc1);
    acc0 = wmma_bf16(bmf0[1], uf1, acc0);
    acc1 = wmma_bf16(bmf1[1], uf1, acc1);

    // += A @ h_t : two independent accumulation chains over preloaded frags
#pragma unroll
    for (int k = 0; k < 8; ++k) {
      acc0 = wmma_bf16(afr0[k], hf[k], acc0);
      acc1 = wmma_bf16(afr1[k], hf[k], acc1);
    }

    // tanh (hardware TRANS op)
    float t0[8], t1[8];
#pragma unroll
    for (int r = 0; r < 8; ++r) t0[r] = fast_tanh(acc0[r]);
#pragma unroll
    for (int r = 0; r < 8; ++r) t1[r] = fast_tanh(acc1[r]);

    // fused output projection: partial of D . h_next for batch lm
    float p = 0.0f;
#pragma unroll
    for (int r = 0; r < 8; ++r) p += df0[r] * t0[r] + df1[r] * t1[r];
    p += __shfl_xor(p, 16, 32);                  // combine lane halves
    if (hs == 0) atomicAdd(&sOut[t & 1][lm], p); // ds_add_f32

    // store h_next (bf16, one 16B LDS store per tile)
    v8bf hp0, hp1;
#pragma unroll
    for (int r = 0; r < 8; ++r) { hp0[r] = (__bf16)t0[r]; hp1[r] = (__bf16)t1[r]; }
    const int nxt = cur ^ 1;
    *(v8bf*)&hbuf[nxt][lm][mt0 * 16 + hs * 8] = hp0;
    *(v8bf*)&hbuf[nxt][lm][mt1 * 16 + hs * 8] = hp1;

    __syncthreads();  // the ONLY barrier per timestep

    if (tid < 16) {
      const int tb = t & 1;
      float o = sOut[tb][tid];
      sOut[tb][tid] = 0.0f;   // re-arm this buffer for step t+2
      out[(size_t)(bt * 16 + tid) * T_LEN + t] = o + E0;
    }
    cur = nxt;
  }
}

extern "C" void kernel_launch(void* const* d_in, const int* in_sizes, int n_in,
                              void* d_out, int out_size, void* d_ws, size_t ws_size,
                              hipStream_t stream) {
  (void)in_sizes; (void)n_in; (void)out_size; (void)d_ws; (void)ws_size;
  const float* u  = (const float*)d_in[0];
  const float* A  = (const float*)d_in[1];
  const float* Bm = (const float*)d_in[2];
  const float* C  = (const float*)d_in[3];
  const float* D  = (const float*)d_in[4];
  const float* E  = (const float*)d_in[5];
  float* out = (float*)d_out;

  srnn_fused_kernel<<<dim3(B_DIM / 16), dim3(256), 0, stream>>>(u, A, Bm, C, D, E, out);
}
